// PointerDecoder_54374285967372
// MI455X (gfx1250) — matrix operand order repaired and checked by reading, not compiled, not run on hardware
//
#include <hip/hip_runtime.h>
#include <hip/hip_bf16.h>

#define B_    256
#define P_    100
#define H_    512
#define NH_   16
#define HD_   32
#define V_    20
#define N_    200
#define VN_   221
#define VNP_  224                       // VN padded for aligned bf16 rows
#define CSD_  16
#define KIN_  528                       // H + CSD
#define KPAD_ 544                       // KIN padded to multiple of 32
#define SCALE_ 0.17677669529663687f     // 32^-0.5

typedef __bf16 v16bf __attribute__((ext_vector_type(16)));
typedef float  v8f   __attribute__((ext_vector_type(8)));
typedef unsigned short u16x8 __attribute__((ext_vector_type(8)));

union BF16x16 { v16bf v; unsigned short u[16]; u16x8 h[2]; };

__device__ inline unsigned short f2b(float f) {
  unsigned int x = __float_as_uint(f);
  unsigned int r = x + 0x7fffu + ((x >> 16) & 1u);   // round-to-nearest-even
  return (unsigned short)(r >> 16);
}
__device__ inline unsigned int f2b2(float lo, float hi) {
  return (unsigned int)f2b(lo) | ((unsigned int)f2b(hi) << 16);
}

// A fragment (16x32 bf16): lane holds row M=lane&15; per-lane K offsets are two
// contiguous 8-element runs: [g*8, g*8+8) and [16+g*8, 16+g*8+8).  2x 16B loads.
__device__ inline void a_frag_bf(BF16x16& a, const unsigned short* rowp, int k0, int g) {
  a.h[0] = *(const u16x8*)(rowp + k0 + g * 8);
  a.h[1] = *(const u16x8*)(rowp + k0 + 16 + g * 8);
}
// B fragment (32x16 bf16, B stored [k][n]): lane holds column n0+(lane&15); K offsets
// match A: [g*8, g*8+8) and [16+g*8, 16+g*8+8).  Strided scalar loads.
__device__ inline void b_frag_bf(BF16x16& bm, const unsigned short* B, int ldb, int k, int n0) {
  int l = k & 31, kb = k - l, hh = l >> 4;
  const unsigned short* p = B + n0 + (l & 15);
  #pragma unroll
  for (int t = 0; t < 8; ++t) {
    bm.u[t]     = p[(size_t)(kb + hh * 8 + t) * ldb];
    bm.u[8 + t] = p[(size_t)(kb + 16 + hh * 8 + t) * ldb];
  }
}

__device__ inline v8f vzero8() {
  v8f z;
  #pragma unroll
  for (int j = 0; j < 8; ++j) z[j] = 0.f;
  return z;
}
__device__ inline v8f wmma_bf16(const BF16x16& a, const BF16x16& b, v8f c) {
  return __builtin_amdgcn_wmma_f32_16x16x32_bf16(false, a.v, false, b.v,
                                                 (short)0, c, false, false);
}
__device__ inline float grp_max16(float x) {
  x = fmaxf(x, __shfl_xor(x, 1, 32));
  x = fmaxf(x, __shfl_xor(x, 2, 32));
  x = fmaxf(x, __shfl_xor(x, 4, 32));
  x = fmaxf(x, __shfl_xor(x, 8, 32));
  return x;
}
__device__ inline float grp_sum16(float x) {
  x += __shfl_xor(x, 1, 32);
  x += __shfl_xor(x, 2, 32);
  x += __shfl_xor(x, 4, 32);
  x += __shfl_xor(x, 8, 32);
  return x;
}

// ================= pre-pass packing kernels (one-time f32 -> bf16) =================
// qin[row][0..543] = concat(q_global, context_feature) zero-padded
__global__ __launch_bounds__(256)
void pack_qin_kernel(const float* __restrict__ qg, const float* __restrict__ cf,
                     unsigned short* __restrict__ qin) {
  int id = blockIdx.x * 256 + threadIdx.x;
  if (id >= (B_ * P_) * (KPAD_ / 4)) return;
  int row = id / (KPAD_ / 4);
  int k = (id % (KPAD_ / 4)) * 4;
  float4 w;
  if (k < H_)        w = *(const float4*)(qg + (size_t)row * H_ + k);
  else if (k < KIN_) w = *(const float4*)(cf + (size_t)row * CSD_ + (k - H_));
  else               w = make_float4(0.f, 0.f, 0.f, 0.f);
  *(uint2*)(qin + (size_t)row * KPAD_ + k) = make_uint2(f2b2(w.x, w.y), f2b2(w.z, w.w));
}

// Bm[k][n] = (k < kin) ? w[n*kin + k] : 0   (transpose [out,in] -> [k][n])
__global__ __launch_bounds__(256)
void pack_bT_kernel(const float* __restrict__ w, unsigned short* __restrict__ Bm,
                    int kin, int kpad, int nout) {
  int id = blockIdx.x * 256 + threadIdx.x;
  if (id >= kpad * (nout / 4)) return;
  int k = id / (nout / 4);
  int n = (id % (nout / 4)) * 4;
  float x0 = 0.f, x1 = 0.f, x2 = 0.f, x3 = 0.f;
  if (k < kin) {
    x0 = w[(size_t)(n + 0) * kin + k];
    x1 = w[(size_t)(n + 1) * kin + k];
    x2 = w[(size_t)(n + 2) * kin + k];
    x3 = w[(size_t)(n + 3) * kin + k];
  }
  *(uint2*)(Bm + (size_t)k * nout + n) = make_uint2(f2b2(x0, x1), f2b2(x2, x3));
}

// cpb[b*H+k][0..223] = cache_pointer[b][k][n] zero-padded beyond 221
__global__ __launch_bounds__(256)
void pack_cp_kernel(const float* __restrict__ cp, unsigned short* __restrict__ cpb) {
  int id = blockIdx.x * 256 + threadIdx.x;
  if (id >= B_ * H_ * (VNP_ / 4)) return;
  int bk = id / (VNP_ / 4);
  int n = (id % (VNP_ / 4)) * 4;
  const float* src = cp + (size_t)bk * VN_;
  float x0 = (n + 0 < VN_) ? src[n + 0] : 0.f;
  float x1 = (n + 1 < VN_) ? src[n + 1] : 0.f;
  float x2 = (n + 2 < VN_) ? src[n + 2] : 0.f;
  float x3 = (n + 3 < VN_) ? src[n + 3] : 0.f;
  *(uint2*)(cpb + (size_t)bk * VNP_ + n) = make_uint2(f2b2(x0, x1), f2b2(x2, x3));
}

// ================= Kernel 1: Q projection (16x64 per wave, 17 uniform k-steps) =======
__global__ __launch_bounds__(32)
void qproj_kernel(const unsigned short* __restrict__ qin,
                  const unsigned short* __restrict__ Bq,
                  unsigned short* __restrict__ qbf) {
  int mt = blockIdx.x, nt4 = blockIdx.y;
  int lane = threadIdx.x;
  int g = lane >> 4, mr = lane & 15, nc = lane & 15;
  const unsigned short* arow = qin + (size_t)(mt * 16 + mr) * KPAD_;
  v8f acc[4];
  #pragma unroll
  for (int u = 0; u < 4; ++u) acc[u] = vzero8();
  for (int kt = 0; kt < 17; ++kt) {
    int k0 = kt * 32;
    BF16x16 a;
    a_frag_bf(a, arow, k0, g);
    #pragma unroll
    for (int u = 0; u < 4; ++u) {
      BF16x16 bm;
      b_frag_bf(bm, Bq, H_, k0 + lane, nt4 * 64 + u * 16);
      acc[u] = wmma_bf16(a, bm, acc[u]);
    }
  }
  #pragma unroll
  for (int u = 0; u < 4; ++u)
    #pragma unroll
    for (int j = 0; j < 8; ++j) {
      int m = j + (g << 3);
      qbf[(size_t)(mt * 16 + m) * H_ + nt4 * 64 + u * 16 + nc] = f2b(acc[u][j]);
    }
}

// ================= Kernel 2: 4-way attention, one workgroup per (b, head) ============
__global__ __launch_bounds__(224)
void attn_kernel(const unsigned short* __restrict__ q,
                 const float* __restrict__ kg,  const float* __restrict__ vg,
                 const float* __restrict__ kn,  const float* __restrict__ vn,
                 const float* __restrict__ kv,  const float* __restrict__ vv,
                 const float* __restrict__ kvn, const float* __restrict__ vvn,
                 const unsigned char* __restrict__ msk,
                 unsigned short* __restrict__ ctx) {
  __shared__ __align__(16) unsigned short sKT[32][32];    // [hd][key] (transposed K)
  __shared__ __align__(16) unsigned short sV [32][32];    // [key][hd]
  __shared__ __align__(16) unsigned short sW [7][16][32]; // per-wave probs (C->A bounce)

  int bh = blockIdx.x;
  int b = bh / NH_, h = bh % NH_;
  int tid = threadIdx.x;
  int wave = tid >> 5, lane = tid & 31;
  int g = lane >> 4, mr = lane & 15, nc = lane & 15;

  const float* Ks[4] = { kg  + (size_t)bh * VN_      * HD_,
                         kn  + (size_t)bh * (N_ + 1) * HD_,
                         kv  + (size_t)bh * (V_ + 1) * HD_,
                         kvn + (size_t)bh * V_       * HD_ };
  const float* Vs[4] = { vg  + (size_t)bh * VN_      * HD_,
                         vn  + (size_t)bh * (N_ + 1) * HD_,
                         vv  + (size_t)bh * (V_ + 1) * HD_,
                         vvn + (size_t)bh * V_       * HD_ };
  const int lens[4] = { VN_, N_ + 1, V_ + 1, V_ };

  BF16x16 qa;
  {
    int p = wave * 16 + mr;
    if (p < P_) {
      const unsigned short* qrow = q + (size_t)(b * P_ + p) * H_ + h * HD_;
      qa.h[0] = *(const u16x8*)(qrow + g * 8);
      qa.h[1] = *(const u16x8*)(qrow + 16 + g * 8);
    } else {
      #pragma unroll
      for (int t = 0; t < 16; ++t) qa.u[t] = 0;
    }
  }

  v8f ct0 = vzero8(), ct1 = vzero8();

  for (int s = 0; s < 4; ++s) {
    const float* Kp = Ks[s]; const float* Vp = Vs[s];
    int len = lens[s];
    int nch = (len + 31) >> 5;
    v8f cs0 = vzero8(), cs1 = vzero8();
    float rmax[8], rsum[8];
    #pragma unroll
    for (int j = 0; j < 8; ++j) { rmax[j] = -1e30f; rsum[j] = 0.f; }

    for (int c = 0; c < nch; ++c) {
      int nb = c << 5;
      __syncthreads();
      for (int idx = tid; idx < 256; idx += 224) {
        int key = idx >> 3, q4 = (idx & 7) << 2;
        int kgl = nb + key;
        float4 kd = make_float4(0.f, 0.f, 0.f, 0.f);
        float4 vd = make_float4(0.f, 0.f, 0.f, 0.f);
        if (kgl < len) {
          kd = *(const float4*)(Kp + (size_t)kgl * HD_ + q4);
          vd = *(const float4*)(Vp + (size_t)kgl * HD_ + q4);
        }
        *(uint2*)&sV[key][q4] = make_uint2(f2b2(vd.x, vd.y), f2b2(vd.z, vd.w));
        sKT[q4 + 0][key] = f2b(kd.x);
        sKT[q4 + 1][key] = f2b(kd.y);
        sKT[q4 + 2][key] = f2b(kd.z);
        sKT[q4 + 3][key] = f2b(kd.w);
      }
      __syncthreads();

      BF16x16 bk0, bk1;
      #pragma unroll
      for (int t = 0; t < 8; ++t) {
        bk0.u[t] = sKT[g * 8 + t][nc]; bk0.u[8 + t] = sKT[16 + g * 8 + t][nc];
        bk1.u[t] = sKT[g * 8 + t][16 + nc]; bk1.u[8 + t] = sKT[16 + g * 8 + t][16 + nc];
      }
      v8f s0 = wmma_bf16(qa, bk0, vzero8());
      v8f s1 = wmma_bf16(qa, bk1, vzero8());

      float x0[8], x1[8];
      #pragma unroll
      for (int j = 0; j < 8; ++j) {
        float a0 = s0[j] * SCALE_, a1 = s1[j] * SCALE_;
        int m = j + (g << 3);
        int p = wave * 16 + m;
        int k0i = nb + nc, k1i = k0i + 16;
        bool bad0 = (k0i >= len), bad1 = (k1i >= len);
        if (s == 0 && p < P_) {
          const unsigned char* mrow = msk + (size_t)(b * P_ + p) * VN_;
          if (!bad0 && mrow[k0i]) bad0 = true;
          if (!bad1 && mrow[k1i]) bad1 = true;
        }
        x0[j] = bad0 ? -1e30f : a0;
        x1[j] = bad1 ? -1e30f : a1;
      }
      #pragma unroll
      for (int j = 0; j < 8; ++j) {
        float cm = grp_max16(fmaxf(x0[j], x1[j]));
        float nm = fmaxf(rmax[j], cm);
        float sc = __expf(rmax[j] - nm);
        float w0 = __expf(x0[j] - nm);
        float w1 = __expf(x1[j] - nm);
        float rs = grp_sum16(w0 + w1);
        rsum[j] = rsum[j] * sc + rs;
        rmax[j] = nm;
        cs0[j] *= sc; cs1[j] *= sc;
        int m = j + (g << 3);
        sW[wave][m][nc]      = f2b(w0);
        sW[wave][m][16 + nc] = f2b(w1);
      }
      __syncthreads();

      BF16x16 wa, bv0, bv1;
      wa.h[0] = *(const u16x8*)&sW[wave][mr][g * 8];
      wa.h[1] = *(const u16x8*)&sW[wave][mr][16 + g * 8];
      #pragma unroll
      for (int t = 0; t < 8; ++t) {
        bv0.u[t] = sV[g * 8 + t][nc]; bv0.u[8 + t] = sV[16 + g * 8 + t][nc];
        bv1.u[t] = sV[g * 8 + t][16 + nc]; bv1.u[8 + t] = sV[16 + g * 8 + t][16 + nc];
      }
      cs0 = wmma_bf16(wa, bv0, cs0);
      cs1 = wmma_bf16(wa, bv1, cs1);
    }
    #pragma unroll
    for (int j = 0; j < 8; ++j) {
      float inv = 1.0f / rsum[j];
      ct0[j] += cs0[j] * inv;
      ct1[j] += cs1[j] * inv;
    }
  }

  #pragma unroll
  for (int j = 0; j < 8; ++j) {
    int m = j + (g << 3);
    int p = wave * 16 + m;
    if (p < P_) {
      unsigned short* crow = ctx + (size_t)(b * P_ + p) * H_ + h * HD_;
      crow[nc]      = f2b(ct0[j]);
      crow[16 + nc] = f2b(ct1[j]);
    }
  }
}

// ================= Kernel 3: combine (16x64 per wave) ================================
__global__ __launch_bounds__(32)
void combine_kernel(const unsigned short* __restrict__ ctx,
                    const unsigned short* __restrict__ Bc,
                    const float* __restrict__ cb,
                    unsigned short* __restrict__ mha) {
  int mt = blockIdx.x, nt4 = blockIdx.y;
  int lane = threadIdx.x;
  int g = lane >> 4, mr = lane & 15, nc = lane & 15;
  const unsigned short* arow = ctx + (size_t)(mt * 16 + mr) * H_;
  v8f acc[4];
  #pragma unroll
  for (int u = 0; u < 4; ++u) acc[u] = vzero8();
  for (int kt = 0; kt < 16; ++kt) {
    int k0 = kt * 32;
    BF16x16 a;
    a_frag_bf(a, arow, k0, g);
    #pragma unroll
    for (int u = 0; u < 4; ++u) {
      BF16x16 bm;
      b_frag_bf(bm, Bc, H_, k0 + lane, nt4 * 64 + u * 16);
      acc[u] = wmma_bf16(a, bm, acc[u]);
    }
  }
  #pragma unroll
  for (int u = 0; u < 4; ++u) {
    float bias = cb[nt4 * 64 + u * 16 + nc];
    #pragma unroll
    for (int j = 0; j < 8; ++j) {
      int m = j + (g << 3);
      mha[(size_t)(mt * 16 + m) * H_ + nt4 * 64 + u * 16 + nc] = f2b(acc[u][j] + bias);
    }
  }
}

// ================= Kernel 4: logits  mha[b] @ cache_pointer[b] -> f32 out ============
__global__ __launch_bounds__(32)
void logits_kernel(const unsigned short* __restrict__ mha,
                   const unsigned short* __restrict__ cpb,
                   float* __restrict__ out) {
  int b = blockIdx.x, pt = blockIdx.y, ntl = blockIdx.z;
  int lane = threadIdx.x;
  int g = lane >> 4, mr = lane & 15, nc = lane & 15;
  int p = pt * 16 + mr;
  bool pv = (p < P_);
  int n0 = ntl * 16;
  const unsigned short* arow = mha + (size_t)(b * P_ + (pv ? p : 0)) * H_;
  const unsigned short* Bb = cpb + (size_t)b * H_ * VNP_;
  v8f acc = vzero8();
  for (int kt = 0; kt < 16; ++kt) {
    int k0 = kt * 32;
    BF16x16 a;
    if (pv) a_frag_bf(a, arow, k0, g);
    else {
      #pragma unroll
      for (int t = 0; t < 16; ++t) a.u[t] = 0;
    }
    BF16x16 bm;
    b_frag_bf(bm, Bb, VNP_, k0 + lane, n0);   // rows zero-padded beyond 221
    acc = wmma_bf16(a, bm, acc);
  }
  #pragma unroll
  for (int j = 0; j < 8; ++j) {
    int m  = j + (g << 3);
    int pp = pt * 16 + m;
    int n  = n0 + nc;
    if (pp < P_ && n < VN_)
      out[(size_t)(b * P_ + pp) * VN_ + n] = acc[j];
  }
}

extern "C" void kernel_launch(void* const* d_in, const int* in_sizes, int n_in,
                              void* d_out, int out_size, void* d_ws, size_t ws_size,
                              hipStream_t stream) {
  const float* qg  = (const float*)d_in[0];
  const float* kgl = (const float*)d_in[1];
  const float* vgl = (const float*)d_in[2];
  const float* knd = (const float*)d_in[3];
  const float* vnd = (const float*)d_in[4];
  const float* kvh = (const float*)d_in[5];
  const float* vvh = (const float*)d_in[6];
  const float* kvn = (const float*)d_in[7];
  const float* vvn = (const float*)d_in[8];
  const float* cf  = (const float*)d_in[9];
  const float* cp  = (const float*)d_in[10];
  const unsigned char* msk = (const unsigned char*)d_in[11];
  const float* wq  = (const float*)d_in[12];
  const float* cw  = (const float*)d_in[13];
  const float* cb  = (const float*)d_in[14];
  float* out = (float*)d_out;

  // workspace layout (bytes, all 256-aligned); mha aliases qin (qin dead by then)
  char* ws = (char*)d_ws;
  const size_t SZ_QIN = (size_t)B_ * P_ * KPAD_ * 2;   // 27,852,800
  const size_t SZ_ROW = (size_t)B_ * P_ * H_ * 2;      // 26,214,400
  const size_t SZ_BQ  = (size_t)KPAD_ * H_ * 2;        // 557,056
  const size_t SZ_BC  = (size_t)H_ * H_ * 2;           // 524,288
  unsigned short* qin  = (unsigned short*)(ws);
  unsigned short* qbf  = (unsigned short*)(ws + SZ_QIN);
  unsigned short* ctxb = (unsigned short*)(ws + SZ_QIN + SZ_ROW);
  unsigned short* Bq   = (unsigned short*)(ws + SZ_QIN + 2 * SZ_ROW);
  unsigned short* Bc   = (unsigned short*)(ws + SZ_QIN + 2 * SZ_ROW + SZ_BQ);
  unsigned short* cpb  = (unsigned short*)(ws + SZ_QIN + 2 * SZ_ROW + SZ_BQ + SZ_BC);
  unsigned short* mhab = qin;                          // reuse

  // pre-pass packing
  pack_qin_kernel<<<(B_ * P_ * (KPAD_ / 4) + 255) / 256, 256, 0, stream>>>(qg, cf, qin);
  pack_bT_kernel<<<(KPAD_ * (H_ / 4) + 255) / 256, 256, 0, stream>>>(wq, Bq, KIN_, KPAD_, H_);
  pack_bT_kernel<<<(H_ * (H_ / 4) + 255) / 256, 256, 0, stream>>>(cw, Bc, H_, H_, H_);
  pack_cp_kernel<<<(B_ * H_ * (VNP_ / 4) + 255) / 256, 256, 0, stream>>>(cp, cpb);

  // main pipeline
  qproj_kernel<<<dim3((B_ * P_) / 16, H_ / 64), dim3(32), 0, stream>>>(qin, Bq, qbf);
  attn_kernel<<<dim3(B_ * NH_), dim3(224), 0, stream>>>(qbf, kgl, vgl, knd, vnd,
                                                        kvh, vvh, kvn, vvn, msk, ctxb);
  combine_kernel<<<dim3((B_ * P_) / 16, H_ / 64), dim3(32), 0, stream>>>(ctxb, Bc, cb, mhab);
  logits_kernel<<<dim3(B_, (P_ + 15) / 16, (VN_ + 15) / 16), dim3(32), 0, stream>>>(mhab, cpb, out);
}
